// Encoder_67138928771138
// MI455X (gfx1250) — matrix operand order repaired and checked by reading, not compiled, not run on hardware
//
#include <hip/hip_runtime.h>
#include <math.h>

// ---------------------------------------------------------------------------
// CfC/NCP RNN (LSTM + 3 masked CfC layers), persistent-kernel formulation.
// MI455X (gfx1250): wave32, bf16 WMMA 16x16x32, LDS-resident LSTM weights
// staged via async global->LDS DMA, device-wide split-phase barrier between
// recurrence phases.
// ---------------------------------------------------------------------------

typedef __bf16 bf16;
typedef __attribute__((ext_vector_type(16))) __bf16       v16bf;
typedef __attribute__((ext_vector_type(8)))  float        v8f;
typedef __attribute__((ext_vector_type(4)))  unsigned int u32x4;

#define BATCH   128
#define TSTEPS  256
#define NVARS   8
#define INDIM   9
#define LATENT  512
#define G4      2048      // 4*LATENT
#define INTER   216
#define COMMAND 143
#define MOTOR   153

// Padded K (multiple of 32) and O (multiple of 16) per matmul
#define KL  544           // LSTM:  [xt(9) | pad->32 | h(512)]
#define KP0 256           // layer0 K = 9+216 = 225
#define KP1 384           // layer1 K = 216+143 = 359
#define KP2 320           // layer2 K = 143+153 = 296
#define OP0 224           // 216 -> 224
#define OP1 144           // 143 -> 144
#define OP2 160           // 153 -> 160
#define OT0 14
#define OT1 9
#define OT2 10

#define NBLK 32
#define TPB  256          // 8 wave32s per block; 256 waves total

struct RnnParams {
  const float* x;
  const float* dt;
  const float* lstm_bi;
  const bf16*  WL;                  // packed LSTM weights (2048 x 544) bf16
  const bf16*  w1[3]; const bf16* w2[3]; const bf16* wa[3]; const bf16* wb[3];
  const float* b1[3]; const float* b2[3]; const float* ba[3]; const float* bb[3];
  bf16* xcl; bf16* xc0; bf16* xc1; bf16* xc2;   // activation buffers (bf16)
  float* C;  float* H;                           // LSTM cell + concat hidden (f32)
  unsigned* bar;
  float* out;
};

__device__ __forceinline__ float sigm(float v) { return 1.0f / (1.0f + __expf(-v)); }

// A fragment: 16x32 bf16, M = batch rows (row-major, K contiguous).
// Lane L: m = L&15, kh = (L>>4)*16 -> two contiguous 16B loads (K kh..kh+15).
__device__ __forceinline__ v16bf frag_a(const bf16* xc, int ld, int mbase, int kbase, int lane) {
  int m  = lane & 15;
  int kh = (lane >> 4) << 4;
  const bf16* p = xc + (size_t)(mbase + m) * ld + kbase + kh;
  v16bf f;
  ((u32x4*)&f)[0] = *(const u32x4*)p;
  ((u32x4*)&f)[1] = *(const u32x4*)(p + 8);
  return f;
}

// B fragment: 32x16 bf16 = Wt tile; B[k][n] = W[n][k], W row-major (O x Kpad).
// Lane L: n = L&15, koff = (L>>4)*16 -> two contiguous 16B loads (K koff..koff+15).
__device__ __forceinline__ v16bf frag_b(const bf16* wmat, int ld, int nbase, int kbase, int lane) {
  int n    = lane & 15;
  int koff = (lane >> 4) << 4;
  const bf16* p = wmat + (size_t)(nbase + n) * ld + kbase + koff;
  v16bf f;
  ((u32x4*)&f)[0] = *(const u32x4*)p;
  ((u32x4*)&f)[1] = *(const u32x4*)(p + 8);
  return f;
}

#define WMMA_BF16(a, b, c) \
  __builtin_amdgcn_wmma_f32_16x16x32_bf16(false, (a), false, (b), (short)0, (c), false, false)

// Device-wide split-phase barrier: monotonically increasing counter, no reset.
__device__ __forceinline__ void grid_sync(unsigned* bar, unsigned& target) {
  __syncthreads();
  __threadfence();
  target += NBLK;
  if (threadIdx.x == 0) {
    atomicAdd(bar, 1u);
    while (atomicAdd(bar, 0u) < target) {
      __builtin_amdgcn_s_sleep(2);
    }
  }
  __syncthreads();
}

// One CfC layer: out = tanh(XC*W1t+b1)*(1-ti) + ti*tanh(XC*W2t+b2),
// ti = sigmoid((XC*WBt+bb) - (XC*WAt+ba)). 4 accumulators share one A fragment.
__device__ void cfc_phase(int wave, int lane,
                          const bf16* xc, int ldk, int kchunks,
                          const bf16* pw1, const bf16* pw2, const bf16* pwa, const bf16* pwb,
                          const float* pb1, const float* pb2, const float* pba, const float* pbb,
                          int O, int otiles,
                          bf16* nxt, int ldnxt,          // next layer's input slot (col 0)
                          bf16* xcl, int xcl_off,        // LSTM xc hidden slot
                          float* H, int hoff) {
  if (wave >= 8 * otiles) return;
  int mt = wave / otiles;
  int ot = wave - mt * otiles;
  v8f z = {0.f, 0.f, 0.f, 0.f, 0.f, 0.f, 0.f, 0.f};
  v8f a1 = z, a2 = z, aa = z, ab = z;
  for (int kc = 0; kc < kchunks; ++kc) {
    int kb = kc * 32;
    if (kc + 1 < kchunks) {
      // Warm L2/WGP$ for the next K-chunk of the ff1 weights (global_prefetch_b8)
      __builtin_prefetch(pw1 + (size_t)(ot * 16 + (lane & 15)) * ldk + (kc + 1) * 32, 0, 1);
    }
    v16bf af = frag_a(xc, ldk, mt * 16, kb, lane);
    a1 = WMMA_BF16(af, frag_b(pw1, ldk, ot * 16, kb, lane), a1);
    a2 = WMMA_BF16(af, frag_b(pw2, ldk, ot * 16, kb, lane), a2);
    aa = WMMA_BF16(af, frag_b(pwa, ldk, ot * 16, kb, lane), aa);
    ab = WMMA_BF16(af, frag_b(pwb, ldk, ot * 16, kb, lane), ab);
  }
  int n  = lane & 15;
  int hi = lane >> 4;
  int j  = ot * 16 + n;
  if (j < O) {
    float c1 = pb1[j], c2 = pb2[j], ca = pba[j], cb = pbb[j];
    for (int v = 0; v < 8; ++v) {
      int m = mt * 16 + v + 8 * hi;
      float f1 = tanhf(a1[v] + c1);
      float f2 = tanhf(a2[v] + c2);
      float ti = sigm((ab[v] + cb) - (aa[v] + ca));
      float o  = f1 * (1.0f - ti) + ti * f2;
      if (nxt) nxt[(size_t)m * ldnxt + j] = (bf16)o;
      xcl[(size_t)m * KL + 32 + xcl_off + j] = (bf16)o;
      H[(size_t)m * LATENT + hoff + j] = o;
    }
  }
}

__global__ __launch_bounds__(TPB) void cfc_rnn(RnnParams p) {
  // Block-resident LSTM weight slice: 4 gates x 16 cols x 544 K = 69632 B in LDS
  __shared__ bf16 lds_wl[4 * 16 * KL];
  const int lane = threadIdx.x & 31;
  const int warp = threadIdx.x >> 5;
  const int wave = blockIdx.x * (TPB / 32) + warp;
  const int jt   = blockIdx.x;       // this block's LSTM column tile (0..31)

  // ---- Stage this block's LSTM weight slice into LDS via async DMA --------
  // Slice rows: WL[g*512 + jt*16 + r][:], g=0..3, r=0..15; row pitch = KL*2 =
  // 1088 B = 68 x 16B chunks. Each lane moves one 16B chunk per issue:
  // global_load_async_to_lds_b128 (GVS: saddr64 + vaddr32), tracked by
  // ASYNCcnt, then s_wait_asynccnt 0 + barrier before ds_load consumers.
  {
    const unsigned lds_base = (unsigned)(size_t)(&lds_wl[0]);
    const bf16* gbase = p.WL;
    const int nchunks = (4 * 16 * KL * 2) / 16;    // 4352
    for (int c = threadIdx.x; c < nchunks; c += TPB) {
      int rowIdx = c / 68;                 // 0..63  (g*16 + r)
      int colb   = (c - rowIdx * 68) * 16; // byte offset within row
      int g = rowIdx >> 4;
      int r = rowIdx & 15;
      unsigned voff = (unsigned)((g * LATENT + jt * 16 + r) * (KL * 2) + colb);
      unsigned ldsa = lds_base + (unsigned)(c * 16);
      asm volatile("global_load_async_to_lds_b128 %0, %1, %2"
                   :: "v"(ldsa), "v"(voff), "s"(gbase)
                   : "memory");
    }
    asm volatile("s_wait_asynccnt 0x0" ::: "memory");
  }
  __syncthreads();

  unsigned target = 0;
  for (int t = 0; t < TSTEPS; ++t) {
    // -------- phase 1: LSTM (all 256 waves; 4 gate accumulators per wave) ---
    {
      int mt = warp;  // batch tile 0..7
      v8f z = {0.f, 0.f, 0.f, 0.f, 0.f, 0.f, 0.f, 0.f};
      v8f ac[4] = {z, z, z, z};
      for (int kc = 0; kc < KL / 32; ++kc) {
        int kb = kc * 32;
        v16bf af = frag_a(p.xcl, KL, mt * 16, kb, lane);
        int n    = lane & 15;
        int koff = (lane >> 4) << 4;
        for (int g = 0; g < 4; ++g) {
          const bf16* bp = &lds_wl[(size_t)(g * 16 + n) * KL + kb + koff];
          v16bf bfr;
          ((u32x4*)&bfr)[0] = *(const u32x4*)bp;        // ds_load_b128
          ((u32x4*)&bfr)[1] = *(const u32x4*)(bp + 8);  // ds_load_b128
          ac[g] = WMMA_BF16(af, bfr, ac[g]);
        }
      }
      int n  = lane & 15;
      int hi = lane >> 4;
      int j  = jt * 16 + n;
      float bi0 = p.lstm_bi[j];
      float bi1 = p.lstm_bi[512 + j];
      float bi2 = p.lstm_bi[1024 + j];
      float bi3 = p.lstm_bi[1536 + j];
      for (int v = 0; v < 8; ++v) {
        int m = mt * 16 + v + 8 * hi;
        float zi = ac[0][v] + bi0;
        float zg = ac[1][v] + bi1;
        float zf = ac[2][v] + bi2;
        float zo = ac[3][v] + bi3;
        float cp = p.C[(size_t)m * LATENT + j];
        float cn = cp * sigm(zf + 1.0f) + tanhf(zi) * sigm(zg);
        float h  = tanhf(cn) * sigm(zo);
        p.C[(size_t)m * LATENT + j] = cn;
        if (j < INTER)                 p.xc0[(size_t)m * KP0 + INDIM + j] = (bf16)h;
        else if (j < INTER + COMMAND)  p.xc1[(size_t)m * KP1 + j]         = (bf16)h;
        else                           p.xc2[(size_t)m * KP2 + (COMMAND + j - (INTER + COMMAND))] = (bf16)h;
      }
    }
    grid_sync(p.bar, target);

    // -------- phase 2: CfC layer 0 (112 waves) ------------------------------
    cfc_phase(wave, lane, p.xc0, KP0, KP0 / 32,
              p.w1[0], p.w2[0], p.wa[0], p.wb[0],
              p.b1[0], p.b2[0], p.ba[0], p.bb[0],
              INTER, OT0, p.xc1, KP1, p.xcl, 0, p.H, 0);
    grid_sync(p.bar, target);

    // -------- phase 3: CfC layer 1 (72 waves) -------------------------------
    cfc_phase(wave, lane, p.xc1, KP1, KP1 / 32,
              p.w1[1], p.w2[1], p.wa[1], p.wb[1],
              p.b1[1], p.b2[1], p.ba[1], p.bb[1],
              COMMAND, OT1, p.xc2, KP2, p.xcl, INTER, p.H, INTER);
    grid_sync(p.bar, target);

    // -------- phase 4: CfC layer 2 (80 waves) + stage next xt ---------------
    cfc_phase(wave, lane, p.xc2, KP2, KP2 / 32,
              p.w1[2], p.w2[2], p.wa[2], p.wb[2],
              p.b1[2], p.b2[2], p.ba[2], p.bb[2],
              MOTOR, OT2, (bf16*)nullptr, 0, p.xcl, INTER + COMMAND, p.H, INTER + COMMAND);
    if (t + 1 < TSTEPS) {
      int tid = blockIdx.x * TPB + threadIdx.x;
      if (tid < BATCH * INDIM) {
        int b = tid / INDIM;
        int v = tid - b * INDIM;
        float val = (v < NVARS) ? p.x[(size_t)b * TSTEPS * NVARS + (size_t)(t + 1) * NVARS + v]
                                : p.dt[(size_t)b * TSTEPS + (t + 1)];
        p.xcl[(size_t)b * KL + v]  = (bf16)val;
        p.xc0[(size_t)b * KP0 + v] = (bf16)val;
      }
    }
    grid_sync(p.bar, target);
  }

  // -------- emit (h, c) -----------------------------------------------------
  int tid = blockIdx.x * TPB + threadIdx.x;
  for (int i = tid; i < BATCH * LATENT; i += NBLK * TPB) {
    p.out[i] = p.H[i];
    p.out[BATCH * LATENT + i] = p.C[i];
  }
}

// ----------------------------- prep kernels --------------------------------

__global__ void pack_lstm_k(const float* wi, const float* wr, bf16* wl) {
  int i = blockIdx.x * blockDim.x + threadIdx.x;
  if (i >= G4 * KL) return;
  int n = i / KL, k = i - n * KL;
  float v = 0.0f;
  if (k < INDIM)      v = wi[n * INDIM + k];
  else if (k >= 32)   v = wr[(size_t)n * LATENT + (k - 32)];
  wl[i] = (bf16)v;
}

__global__ void pack_w_k(const float* w, const int* mask, bf16* dst,
                         int O, int K, int Opad, int Kpad) {
  int i = blockIdx.x * blockDim.x + threadIdx.x;
  if (i >= Opad * Kpad) return;
  int o = i / Kpad, k = i - o * Kpad;
  float v = 0.0f;
  if (o < O && k < K) {
    v = w[(size_t)o * K + k];
    if (mask) v *= (float)mask[(size_t)o * K + k];
  }
  dst[i] = (bf16)v;
}

__global__ void pack_b_k(const float* b, float* dst, int O, int Opad) {
  int i = blockIdx.x * blockDim.x + threadIdx.x;
  if (i >= Opad) return;
  dst[i] = (i < O) ? b[i] : 0.0f;
}

__global__ void init_k(bf16* xcl, bf16* xc0, bf16* xc1, bf16* xc2,
                       float* C, float* H, unsigned* bar,
                       const float* x, const float* dt) {
  int i = blockIdx.x * blockDim.x + threadIdx.x;
  int stride = gridDim.x * blockDim.x;
  for (int k = i; k < BATCH * KL; k += stride) {
    int b = k / KL, c = k - b * KL;
    float v = 0.0f;
    if (c < INDIM) v = (c < NVARS) ? x[(size_t)b * TSTEPS * NVARS + c] : dt[(size_t)b * TSTEPS];
    xcl[k] = (bf16)v;
  }
  for (int k = i; k < BATCH * KP0; k += stride) {
    int b = k / KP0, c = k - b * KP0;
    float v = 0.0f;
    if (c < INDIM) v = (c < NVARS) ? x[(size_t)b * TSTEPS * NVARS + c] : dt[(size_t)b * TSTEPS];
    xc0[k] = (bf16)v;
  }
  for (int k = i; k < BATCH * KP1; k += stride) xc1[k] = (bf16)0.0f;
  for (int k = i; k < BATCH * KP2; k += stride) xc2[k] = (bf16)0.0f;
  for (int k = i; k < BATCH * LATENT; k += stride) { C[k] = 0.0f; H[k] = 0.0f; }
  if (i == 0) *bar = 0u;
}

// ----------------------------- host entry ----------------------------------

extern "C" void kernel_launch(void* const* d_in, const int* in_sizes, int n_in,
                              void* d_out, int out_size, void* d_ws, size_t ws_size,
                              hipStream_t stream) {
  (void)in_sizes; (void)n_in; (void)out_size; (void)ws_size;

  const float* x   = (const float*)d_in[0];
  const float* dtp = (const float*)d_in[1];
  const float* lwi = (const float*)d_in[2];
  const float* lwr = (const float*)d_in[3];
  const float* lbi = (const float*)d_in[4];
  const float* w[3][4];
  const float* bv[3][4];
  for (int l = 0; l < 3; ++l)
    for (int q = 0; q < 4; ++q) {
      w[l][q]  = (const float*)d_in[5 + l * 8 + q * 2];
      bv[l][q] = (const float*)d_in[5 + l * 8 + q * 2 + 1];
    }
  const int* msk[3] = {(const int*)d_in[29], (const int*)d_in[30], (const int*)d_in[31]};

  const int Od[3]  = {INTER, COMMAND, MOTOR};
  const int Kd[3]  = {INDIM + INTER, INTER + COMMAND, COMMAND + MOTOR};
  const int OdP[3] = {OP0, OP1, OP2};
  const int KdP[3] = {KP0, KP1, KP2};

  char*  base = (char*)d_ws;
  size_t off  = 0;
  auto carve = [&](size_t bytes) -> void* {
    void* p = base + off;
    off = (off + bytes + 255) & ~(size_t)255;
    return p;
  };

  bf16* WL = (bf16*)carve((size_t)G4 * KL * sizeof(bf16));
  bf16*  Wp[3][4];
  float* Bp[3][4];
  for (int l = 0; l < 3; ++l)
    for (int q = 0; q < 4; ++q) {
      Wp[l][q] = (bf16*)carve((size_t)OdP[l] * KdP[l] * sizeof(bf16));
      Bp[l][q] = (float*)carve((size_t)OdP[l] * sizeof(float));
    }
  bf16* xcl = (bf16*)carve((size_t)BATCH * KL  * sizeof(bf16));
  bf16* xc0 = (bf16*)carve((size_t)BATCH * KP0 * sizeof(bf16));
  bf16* xc1 = (bf16*)carve((size_t)BATCH * KP1 * sizeof(bf16));
  bf16* xc2 = (bf16*)carve((size_t)BATCH * KP2 * sizeof(bf16));
  float* C  = (float*)carve((size_t)BATCH * LATENT * sizeof(float));
  float* H  = (float*)carve((size_t)BATCH * LATENT * sizeof(float));
  unsigned* bar = (unsigned*)carve(256);

  pack_lstm_k<<<(G4 * KL + TPB - 1) / TPB, TPB, 0, stream>>>(lwi, lwr, WL);
  for (int l = 0; l < 3; ++l)
    for (int q = 0; q < 4; ++q) {
      int elems = OdP[l] * KdP[l];
      pack_w_k<<<(elems + TPB - 1) / TPB, TPB, 0, stream>>>(
          w[l][q], (q < 2) ? msk[l] : nullptr, Wp[l][q], Od[l], Kd[l], OdP[l], KdP[l]);
      pack_b_k<<<1, TPB, 0, stream>>>(bv[l][q], Bp[l][q], Od[l], OdP[l]);
    }
  init_k<<<128, TPB, 0, stream>>>(xcl, xc0, xc1, xc2, C, H, bar, x, dtp);

  RnnParams prm;
  prm.x = x; prm.dt = dtp; prm.lstm_bi = lbi; prm.WL = WL;
  for (int l = 0; l < 3; ++l) {
    prm.w1[l] = Wp[l][0]; prm.w2[l] = Wp[l][1]; prm.wa[l] = Wp[l][2]; prm.wb[l] = Wp[l][3];
    prm.b1[l] = Bp[l][0]; prm.b2[l] = Bp[l][1]; prm.ba[l] = Bp[l][2]; prm.bb[l] = Bp[l][3];
  }
  prm.xcl = xcl; prm.xc0 = xc0; prm.xc1 = xc1; prm.xc2 = xc2;
  prm.C = C; prm.H = H; prm.bar = bar;
  prm.out = (float*)d_out;

  cfc_rnn<<<NBLK, TPB, 0, stream>>>(prm);
}
